// CausalAttention_12661563588859
// MI455X (gfx1250) — matrix operand order repaired and checked by reading, not compiled, not run on hardware
//
#include <hip/hip_runtime.h>
#include <hip/hip_bf16.h>

// ---------------------------------------------------------------------------
// CausalAttention (B=4,S=1024,H=2048,NH=32,NKV=8,HD=64) for gfx1250 (MI455X)
// Matrix math on v_wmma_f32_16x16x32_f16 (wave32 WMMA). fp32 I/O per ref.
// aw (537MB) streamed with non-temporal hints to protect L2 reuse of
// mask / v / attn / weights.
// ---------------------------------------------------------------------------

typedef __attribute__((ext_vector_type(16))) _Float16 v16h;
typedef __attribute__((ext_vector_type(2)))  _Float16 v2h;
typedef __attribute__((ext_vector_type(8)))  float    v8f;

constexpr int B_  = 4;
constexpr int S_  = 1024;
constexpr int H_  = 2048;
constexpr int NH_ = 32;
constexpr int NKV_ = 8;
constexpr int HD_ = 64;
constexpr int NREP_ = NH_ / NKV_;   // 4

static __device__ __forceinline__ v8f wmma_f16(const v16h& a, const v16h& b, v8f c) {
    return __builtin_amdgcn_wmma_f32_16x16x32_f16(false, a, false, b, (short)0, c,
                                                  false, false);
}

// ---------------------------------------------------------------------------
// Generic GEMM: C[M,N](f32) = A[M,K](f32) x B[K,N](f32), math in f16 WMMA.
// Block tile 128(M) x 64(N), k-step 32, 8 waves (4x2), wave tile 32x32.
// MODE 0: unbatched. MODE 1: AV (batched over z = b*NH + h, GQA on B,
//         C scattered into [B,S,NH*HD]); A (=aw) is a 537MB single-use
//         stream -> non-temporal loads.
// ---------------------------------------------------------------------------
template <int MODE>
__global__ __launch_bounds__(256) void gemm_f16wmma_kernel(
    const float* __restrict__ A, const float* __restrict__ B,
    float* __restrict__ C, int M, int N, int K, int lda, int ldb, int ldc) {

    if (MODE == 1) {
        const int z = blockIdx.z;
        const int bb = z / NH_;
        const int hh = z % NH_;
        A += (size_t)z * S_ * S_;                                  // aw[b,h]
        B += (size_t)bb * S_ * (NKV_ * HD_) + (hh / NREP_) * HD_;  // v[b,kvh]
        C += (size_t)bb * S_ * H_ + (size_t)hh * HD_;              // attn[b,:,h*64]
    }

    __shared__ _Float16 As[128 * 32];   // [row][k] row-major, 8 KB
    __shared__ _Float16 Bs[64 * 32];    // [col][k] transposed,  4 KB

    const int tid  = threadIdx.x;
    const int lane = tid & 31;
    const int wave = tid >> 5;
    const int wm   = wave & 3;          // 4 waves in M
    const int wn   = wave >> 2;         // 2 waves in N
    const int m_blk = blockIdx.y * 128;
    const int n_blk = blockIdx.x * 64;

    const int lrow = lane & 15;
    const int ka   = (lane < 16) ? 0 : 8;    // unused; A-frag also uses kb2
    const int kb2  = (lane < 16) ? 0 : 16;   // B-frag K sub-offset

    v8f acc[2][2] = {};

    for (int kb = 0; kb < K; kb += 32) {
        // --- stage A tile (128x32), 2 threads per row, f32 -> packed f16 ---
        {
            const int row  = tid >> 1;
            const int coff = (tid & 1) * 16;
            const float* src = A + (size_t)(m_blk + row) * lda + kb + coff;
            v2h* dst = (v2h*)&As[row * 32 + coff];
#pragma unroll
            for (int j = 0; j < 8; ++j) {
                float f0, f1;
                if (MODE == 1) {
                    f0 = __builtin_nontemporal_load(&src[2 * j]);
                    f1 = __builtin_nontemporal_load(&src[2 * j + 1]);
                } else {
                    f0 = src[2 * j];
                    f1 = src[2 * j + 1];
                }
                v2h p; p.x = (_Float16)f0; p.y = (_Float16)f1;
                dst[j] = p;                       // 32-bit LDS store
            }
            if (kb + 32 < K) __builtin_prefetch((const void*)(src + 32), 0, 0);
        }
        // --- stage B tile (32x64) transposed into Bs[col][k] ---
        {
            const int kk = tid >> 3;
            const int cg = (tid & 7) * 8;
            const float* src = B + (size_t)(kb + kk) * ldb + n_blk + cg;
#pragma unroll
            for (int j = 0; j < 8; ++j) Bs[(cg + j) * 32 + kk] = (_Float16)src[j];
            if (kb + 32 < K) __builtin_prefetch((const void*)(src + 32 * ldb), 0, 0);
        }
        __syncthreads();

        // --- fragments (contiguous k order: k = 16*(lane/16) + s for A and B) ---
        v16h afrag[2], bfrag[2];
#pragma unroll
        for (int mi = 0; mi < 2; ++mi) {
            const _Float16* ap = &As[(wm * 32 + mi * 16 + lrow) * 32];
#pragma unroll
            for (int j = 0; j < 16; ++j) afrag[mi][j] = ap[kb2 + j];
        }
#pragma unroll
        for (int ni = 0; ni < 2; ++ni) {
            const _Float16* bp = &Bs[(wn * 32 + ni * 16 + lrow) * 32 + kb2];
#pragma unroll
            for (int j = 0; j < 16; ++j) bfrag[ni][j] = bp[j];
        }
#pragma unroll
        for (int mi = 0; mi < 2; ++mi)
#pragma unroll
            for (int ni = 0; ni < 2; ++ni)
                acc[mi][ni] = wmma_f16(afrag[mi], bfrag[ni], acc[mi][ni]);

        __syncthreads();
    }

    // --- epilogue: 16x16 f32 C layout (VGPR r -> row r / r+8 by lane half) ---
    const int col_l = lane & 15;
    const int rbase = (lane < 16) ? 0 : 8;
#pragma unroll
    for (int mi = 0; mi < 2; ++mi)
#pragma unroll
        for (int ni = 0; ni < 2; ++ni) {
            const int col = n_blk + wn * 32 + ni * 16 + col_l;
#pragma unroll
            for (int r = 0; r < 8; ++r) {
                const int row = m_blk + wm * 32 + mi * 16 + rbase + r;
                C[(size_t)row * ldc + col] = acc[mi][ni][r];
            }
        }
}

// ---------------------------------------------------------------------------
// RoPE in place on q_f32 [B*S, NH*HD] and k_f32 [B*S, NKV*HD].
// One thread per rotation pair (d, d+32); cos/sin repeat halves so
// c(d) == c(d+32), but index both per the reference layout.
// ---------------------------------------------------------------------------
__global__ __launch_bounds__(256) void rope_kernel(
    float* __restrict__ q, float* __restrict__ k,
    const float* __restrict__ cosp, const float* __restrict__ sinp) {

    const int n_q = B_ * S_ * NH_ * (HD_ / 2);
    const int n_k = B_ * S_ * NKV_ * (HD_ / 2);
    int idx = blockIdx.x * blockDim.x + threadIdx.x;

    float* base;
    int d, bs;
    if (idx < n_q) {
        d = idx & 31; idx >>= 5;
        const int hh = idx % NH_;  idx /= NH_;
        bs = idx;                                   // b*S + s
        base = q + (size_t)bs * H_ + hh * HD_;
    } else if (idx < n_q + n_k) {
        idx -= n_q;
        d = idx & 31; idx >>= 5;
        const int hh = idx % NKV_; idx /= NKV_;
        bs = idx;
        base = k + (size_t)bs * (NKV_ * HD_) + hh * HD_;
    } else {
        return;
    }
    const float c0 = cosp[(size_t)bs * HD_ + d];
    const float s0 = sinp[(size_t)bs * HD_ + d];
    const float c1 = cosp[(size_t)bs * HD_ + d + 32];
    const float s1 = sinp[(size_t)bs * HD_ + d + 32];
    const float v0 = base[d];
    const float v1 = base[d + 32];
    base[d]      = v0 * c0 - v1 * s0;
    base[d + 32] = v1 * c1 + v0 * s1;
}

// ---------------------------------------------------------------------------
// Scores: aw_raw[b,h,m,n] = (q.k^T)*HD^-0.5 + mask. One 16x16 WMMA tile per
// wave, K=64 (two k-steps). q,k read straight from HBM (f32->f16 in regs);
// B-frag columns are k rows, so loads are contiguous. aw written NT (stream);
// mask kept RT so the 4MB/batch slice stays L2-resident across 32 heads.
// grid = (S/128, S/16, B*NH), block = 256 (8 waves).
// ---------------------------------------------------------------------------
__global__ __launch_bounds__(256) void scores_kernel(
    const float* __restrict__ q, const float* __restrict__ k,
    const float* __restrict__ mask, float* __restrict__ aw) {

    const int z   = blockIdx.z;        // b*NH + h
    const int bb  = z / NH_;
    const int hh  = z % NH_;
    const int kvh = hh / NREP_;

    const int lane = threadIdx.x & 31;
    const int wave = threadIdx.x >> 5;
    const int m0   = blockIdx.y * 16;
    const int n0   = blockIdx.x * 128 + wave * 16;

    const int lrow = lane & 15;
    const int ka   = (lane < 16) ? 0 : 8;
    const int kb2  = (lane < 16) ? 0 : 16;

    v8f acc = {};
#pragma unroll
    for (int kk = 0; kk < HD_; kk += 32) {
        v16h aq, bk;
        const float* qp = q + (size_t)(bb * S_ + m0 + lrow) * H_ + hh * HD_ + kk;
        const float* kp = k + (size_t)(bb * S_ + n0 + lrow) * (NKV_ * HD_)
                            + kvh * HD_ + kk + kb2;
#pragma unroll
        for (int j = 0; j < 16; ++j) aq[j] = (_Float16)qp[kb2 + j];
#pragma unroll
        for (int j = 0; j < 16; ++j) bk[j] = (_Float16)kp[j];
        acc = wmma_f16(aq, bk, acc);
    }

    const float scale = 0.125f;   // HD^-0.5 = 1/8
    const int col = n0 + (lane & 15);
    const int rb  = (lane < 16) ? 0 : 8;
#pragma unroll
    for (int r = 0; r < 8; ++r) {
        const int row = m0 + rb + r;
        const float v = acc[r] * scale + mask[((size_t)bb * S_ + row) * S_ + col];
        __builtin_nontemporal_store(v, &aw[((size_t)z * S_ + row) * S_ + col]);
    }
}

// ---------------------------------------------------------------------------
// Row softmax in place on aw: one block per row of 1024, 4 elems/thread.
// Pure streaming pass -> NT loads and stores.
// ---------------------------------------------------------------------------
__global__ __launch_bounds__(256) void softmax_kernel(float* __restrict__ aw) {
    __shared__ float red[256];
    float* p = aw + (size_t)blockIdx.x * S_;
    const int tid = threadIdx.x;

    float local[4];
    float m = -3.4e38f;
#pragma unroll
    for (int j = 0; j < 4; ++j) {
        local[j] = __builtin_nontemporal_load(&p[tid + j * 256]);
        m = fmaxf(m, local[j]);
    }
    red[tid] = m;
    __syncthreads();
    for (int s = 128; s > 0; s >>= 1) {
        if (tid < s) red[tid] = fmaxf(red[tid], red[tid + s]);
        __syncthreads();
    }
    const float rmax = red[0];
    __syncthreads();

    float sum = 0.f;
#pragma unroll
    for (int j = 0; j < 4; ++j) {
        local[j] = __expf(local[j] - rmax);
        sum += local[j];
    }
    red[tid] = sum;
    __syncthreads();
    for (int s = 128; s > 0; s >>= 1) {
        if (tid < s) red[tid] += red[tid + s];
        __syncthreads();
    }
    const float inv = 1.0f / red[0];
#pragma unroll
    for (int j = 0; j < 4; ++j)
        __builtin_nontemporal_store(local[j] * inv, &p[tid + j * 256]);
}

// ---------------------------------------------------------------------------
extern "C" void kernel_launch(void* const* d_in, const int* in_sizes, int n_in,
                              void* d_out, int out_size, void* d_ws, size_t ws_size,
                              hipStream_t stream) {
    const float* x    = (const float*)d_in[0];
    const float* cosp = (const float*)d_in[1];
    const float* sinp = (const float*)d_in[2];
    const float* mask = (const float*)d_in[3];
    const float* Wq   = (const float*)d_in[4];
    const float* Wk   = (const float*)d_in[5];
    const float* Wv   = (const float*)d_in[6];
    const float* Wo   = (const float*)d_in[7];

    const int MS = B_ * S_;            // 4096 GEMM rows
    float* q_f32 = (float*)d_ws;                      // [4096, 2048]
    float* k_f32 = q_f32 + (size_t)MS * H_;           // [4096,  512]
    float* v_f32 = k_f32 + (size_t)MS * (NKV_ * HD_); // [4096,  512]
    float* attn  = v_f32 + (size_t)MS * (NKV_ * HD_); // [4096, 2048]

    float* out = (float*)d_out;                       // [B,S,H]
    float* aw  = out + (size_t)MS * H_;               // [B,NH,S,S]

    dim3 blk(256);

    // q/k/v projections
    gemm_f16wmma_kernel<0><<<dim3(H_ / 64, MS / 128, 1), blk, 0, stream>>>(
        x, Wq, q_f32, MS, H_, H_, H_, H_, H_);
    gemm_f16wmma_kernel<0><<<dim3((NKV_ * HD_) / 64, MS / 128, 1), blk, 0, stream>>>(
        x, Wk, k_f32, MS, NKV_ * HD_, H_, H_, NKV_ * HD_, NKV_ * HD_);
    gemm_f16wmma_kernel<0><<<dim3((NKV_ * HD_) / 64, MS / 128, 1), blk, 0, stream>>>(
        x, Wv, v_f32, MS, NKV_ * HD_, H_, H_, NKV_ * HD_, NKV_ * HD_);

    // RoPE on q and k (in place)
    const int rope_threads = B_ * S_ * (NH_ + NKV_) * (HD_ / 2);
    rope_kernel<<<dim3((rope_threads + 255) / 256), blk, 0, stream>>>(
        q_f32, k_f32, cosp, sinp);

    // scores -> aw region (scaled + masked, pre-softmax)
    scores_kernel<<<dim3(S_ / 128, S_ / 16, B_ * NH_), blk, 0, stream>>>(
        q_f32, k_f32, mask, aw);

    // softmax in place
    softmax_kernel<<<dim3(B_ * NH_ * S_), blk, 0, stream>>>(aw);

    // AV: batched over (b,h), GQA on v
    gemm_f16wmma_kernel<1><<<dim3(HD_ / 64, S_ / 128, B_ * NH_), blk, 0, stream>>>(
        aw, v_f32, attn, S_, HD_, S_, S_, NKV_ * HD_, H_);

    // output projection
    gemm_f16wmma_kernel<0><<<dim3(H_ / 64, MS / 128, 1), blk, 0, stream>>>(
        attn, Wo, out, MS, H_, H_, H_, H_, H_);
}
